// BeaconGPT_34660386079174
// MI455X (gfx1250) — matrix operand order repaired and checked by reading, not compiled, not run on hardware
//
#include <hip/hip_runtime.h>
#include <cstddef>
#include <cstdint>

// ---------------------------------------------------------------------------
// BeaconGPT forward on gfx1250 (MI455X):
//   bf16 WMMA GEMMs fed by Tensor-Data-Mover (TDM) double-buffered LDS tiles,
//   flash attention with online softmax, weights pre-converted to bf16.
// ---------------------------------------------------------------------------

#define SEQ   4096
#define HID   1024
#define NH    8
#define HD    128           // head dim
#define VOCAB 32000
#define NLAYER 4
#define RMSEPS 1e-5f
#define ATT_SCALE 0.08838834764831845f   // 1/sqrt(128)

typedef __bf16 bf16;
typedef __attribute__((ext_vector_type(8)))  float v8f;
typedef __attribute__((ext_vector_type(16))) bf16  v16bf;
typedef __attribute__((ext_vector_type(8)))  bf16  v8bf;
typedef __attribute__((ext_vector_type(4)))  unsigned int v4u;
typedef __attribute__((ext_vector_type(8)))  int v8i;
typedef __attribute__((ext_vector_type(4)))  int v4i;

// ---------------- WMMA helpers (CDNA5 wave32, 16x16x32 bf16 -> f32) --------

__device__ __forceinline__ v8f wmma_bf16(v16bf a, v16bf b, v8f c) {
  // 8 args: (neg_a, A, neg_b, B, c_mod, C, reuse_a, reuse_b)
  return __builtin_amdgcn_wmma_f32_16x16x32_bf16(false, a, false, b,
                                                 (short)0, c, false, false);
}

// A-fragment (16x32 bf16): lane L (g=L>>4) row M=L&15, elem e -> K = ((e>>3)<<4)+8g+(e&7)
__device__ __forceinline__ v16bf frag_a(const bf16* row, int g) {
  union { v16bf v; v8bf h[2]; } u;
  u.h[0] = *(const v8bf*)(row + g * 8);
  u.h[1] = *(const v8bf*)(row + 16 + g * 8);
  return u.v;
}

// B-fragment (32x16 bf16) read from B^T rows (row-major [N][K] tile):
// lane L (g=L>>4) col N=L&15, elem e -> K = ((e>>3)<<4)+8g+(e&7)
__device__ __forceinline__ v16bf frag_b(const bf16* row, int g) {
  union { v16bf v; v8bf h[2]; } u;
  u.h[0] = *(const v8bf*)(row + g * 8);
  u.h[1] = *(const v8bf*)(row + 16 + g * 8);
  return u.v;
}

// ---------------- Tensor Data Mover: 2D bf16 tile -> LDS -------------------
// Builds a D# per CDNA5 ISA and issues TENSOR_LOAD_TO_LDS.
// rows x cols bf16 tile, row stride `stride` elements; lands row-major in LDS.

__device__ __forceinline__ unsigned int lds_off(const void* p) {
  return (unsigned int)(unsigned long long)(uintptr_t)p;  // low 32 = LDS offset
}

__device__ __forceinline__ void tdm_load_2d(unsigned int lds_addr,
                                            const bf16* gsrc,
                                            unsigned int rows, unsigned int cols,
                                            unsigned int stride) {
  const unsigned long long ga = (unsigned long long)(uintptr_t)gsrc;
  v4u g0;
  g0.x = 1u;                                           // count=1 (valid user D#)
  g0.y = lds_addr;                                     // LDS byte address
  g0.z = (unsigned int)ga;                             // global_addr[31:0]
  g0.w = ((unsigned int)(ga >> 32) & 0x01FFFFFFu)      // global_addr[56:32]
         | (2u << 30);                                 // type=2 ("image")
  v8i g1;
  g1[0] = 0x00010000;                                  // data_size=1 (2 bytes)
  g1[1] = (int)(cols << 16);                           // tensor_dim0[15:0]
  g1[2] = (int)((cols >> 16) | (rows << 16));          // dim0 hi | tensor_dim1 lo
  g1[3] = (int)((rows >> 16) | (cols << 16));          // dim1 hi | tile_dim0
  g1[4] = (int)(rows & 0xFFFFu);                       // tile_dim1 (tile_dim2=0)
  g1[5] = (int)stride;                                 // tensor_dim0_stride[31:0]
  g1[6] = 0;
  g1[7] = 0;
  const v4i gz = {0, 0, 0, 0};
#if defined(__clang_major__) && (__clang_major__ >= 23)
  const v8i gz8 = {0, 0, 0, 0, 0, 0, 0, 0};
  __builtin_amdgcn_tensor_load_to_lds(g0, g1, gz, gz, gz8, 0);
#else
  __builtin_amdgcn_tensor_load_to_lds(g0, g1, gz, gz, 0);
#endif
}

// ---------------- fp32 -> bf16 bulk convert (weights / activations) --------

__global__ __launch_bounds__(256) void cvt_kernel(const float* __restrict__ src,
                                                  bf16* __restrict__ dst, int n4) {
  const int i = blockIdx.x * 256 + threadIdx.x;
  if (i >= n4) return;
  __builtin_prefetch(src + (size_t)i * 4 + 8192, 0, 1);  // global_prefetch_b8
  const float4 f = ((const float4*)src)[i];
  dst[(size_t)i * 4 + 0] = (bf16)f.x;
  dst[(size_t)i * 4 + 1] = (bf16)f.y;
  dst[(size_t)i * 4 + 2] = (bf16)f.z;
  dst[(size_t)i * 4 + 3] = (bf16)f.w;
}

// ---------------- RMS norm (optionally with embedding gather) --------------

template <bool EMBED, typename OT>
__global__ __launch_bounds__(256) void rms_kernel(const int* __restrict__ ids,
                                                  const float* __restrict__ src,
                                                  OT* __restrict__ dst) {
  const int s = blockIdx.x, tid = threadIdx.x;
  const float* row = EMBED ? src + (size_t)ids[s] * HID : src + (size_t)s * HID;
  float4 v = *(const float4*)&row[tid * 4];
  float ss = v.x * v.x + v.y * v.y + v.z * v.z + v.w * v.w;
#pragma unroll
  for (int off = 16; off > 0; off >>= 1) ss += __shfl_xor(ss, off, 32);
  __shared__ float red[8];
  if ((tid & 31) == 0) red[tid >> 5] = ss;
  __syncthreads();
  float tot = 0.f;
#pragma unroll
  for (int i = 0; i < 8; ++i) tot += red[i];
  const float sc = rsqrtf(tot * (1.0f / HID) + RMSEPS);
  OT* d = dst + (size_t)s * HID + tid * 4;
  d[0] = (OT)(v.x * sc); d[1] = (OT)(v.y * sc);
  d[2] = (OT)(v.z * sc); d[3] = (OT)(v.w * sc);
}

// ---------------- Tiled GEMM: C[M,N] = A[M,K] @ B[N,K]^T (+res, gelu) ------
// A,B bf16 in global; TDM DMAs 128x32 tiles into double-buffered LDS while
// 8 waves (2x4) run 4x2 WMMA tiles each. Epilogue: +residual / exact GeLU.

template <bool RES, bool GELU, typename CT>
__global__ __launch_bounds__(256) void gemm_kernel(const bf16* __restrict__ A,
                                                   const bf16* __restrict__ B,
                                                   const float* Rsd, CT* C,
                                                   int M, int N, int K) {
  __shared__ bf16 Al[2][128][32];
  __shared__ bf16 Bl[2][128][32];

  const int tid = threadIdx.x;
  const int lane = tid & 31, wave = tid >> 5;
  const int g = lane >> 4, ln = lane & 15;
  const int wm = wave >> 2;  // 0..1 : 64-row slice
  const int wn = wave & 3;   // 0..3 : 32-col slice
  const int n0 = blockIdx.x * 128;
  const int m0 = blockIdx.y * 128;

  v8f acc[4][2];
#pragma unroll
  for (int i = 0; i < 4; ++i)
#pragma unroll
    for (int j = 0; j < 2; ++j)
#pragma unroll
      for (int e = 0; e < 8; ++e) acc[i][j][e] = 0.f;

  if (tid == 0) {  // TDM issued once (ignores EXEC); other waves skip via execz
    tdm_load_2d(lds_off(&Al[0][0][0]), A + (size_t)m0 * K, 128, 32, K);
    tdm_load_2d(lds_off(&Bl[0][0][0]), B + (size_t)n0 * K, 128, 32, K);
  }

  int buf = 0;
  for (int k0 = 0; k0 < K; k0 += 32) {
    if (tid == 0) {
      if (k0 + 32 < K) {  // prefetch next K-slab into the other buffer
        tdm_load_2d(lds_off(&Al[buf ^ 1][0][0]),
                    A + (size_t)m0 * K + k0 + 32, 128, 32, K);
        tdm_load_2d(lds_off(&Bl[buf ^ 1][0][0]),
                    B + (size_t)n0 * K + k0 + 32, 128, 32, K);
        __builtin_amdgcn_s_wait_tensorcnt(2);  // current buffer complete
      } else {
        __builtin_amdgcn_s_wait_tensorcnt(0);
      }
    }
    __syncthreads();

    v16bf bfr[2];
#pragma unroll
    for (int j = 0; j < 2; ++j)
      bfr[j] = frag_b(&Bl[buf][wn * 32 + j * 16 + ln][0], g);
#pragma unroll
    for (int i = 0; i < 4; ++i) {
      const v16bf afr = frag_a(&Al[buf][wm * 64 + i * 16 + ln][0], g);
#pragma unroll
      for (int j = 0; j < 2; ++j) acc[i][j] = wmma_bf16(afr, bfr[j], acc[i][j]);
    }
    __syncthreads();  // done reading buf before it becomes a TDM target again
    buf ^= 1;
  }

#pragma unroll
  for (int i = 0; i < 4; ++i) {
    const int rowb = m0 + wm * 64 + i * 16 + 8 * g;
#pragma unroll
    for (int j = 0; j < 2; ++j) {
      const int col = n0 + wn * 32 + j * 16 + ln;
#pragma unroll
      for (int r = 0; r < 8; ++r) {
        float v = acc[i][j][r];
        const size_t idx = (size_t)(rowb + r) * N + col;
        if (RES) v += Rsd[idx];
        if (GELU) v = 0.5f * v * (1.0f + erff(v * 0.70710678118654752f));
        C[idx] = (CT)v;
      }
    }
  }
}

// ---------------- q/k head-RMS + RoPE + V transpose (bf16 outputs) ---------

__global__ __launch_bounds__(256) void qkprep_kernel(const float* __restrict__ qkv,
                                                     bf16* __restrict__ qb,
                                                     bf16* __restrict__ kb,
                                                     bf16* __restrict__ vt) {
  const int s = blockIdx.x, tid = threadIdx.x;
  const int lane = tid & 31, h = tid >> 5;
  const float* base = qkv + (size_t)s * (3 * HID) + h * HD;
  float4 q = *(const float4*)&base[lane * 4];
  float4 k = *(const float4*)&base[HID + lane * 4];
  const float4 v = *(const float4*)&base[2 * HID + lane * 4];

  float ssq = q.x * q.x + q.y * q.y + q.z * q.z + q.w * q.w;
  float ssk = k.x * k.x + k.y * k.y + k.z * k.z + k.w * k.w;
#pragma unroll
  for (int off = 16; off > 0; off >>= 1) {
    ssq += __shfl_xor(ssq, off, 32);
    ssk += __shfl_xor(ssk, off, 32);
  }
  const float scq = rsqrtf(ssq * (1.0f / HD) + RMSEPS);
  const float sck = rsqrtf(ssk * (1.0f / HD) + RMSEPS);
  q.x *= scq; q.y *= scq; q.z *= scq; q.w *= scq;
  k.x *= sck; k.y *= sck; k.z *= sck; k.w *= sck;

  const float a0 = (float)s * __powf(10000.f, -(float)(2 * lane) * (1.f / 64.f));
  const float a1 = (float)s * __powf(10000.f, -(float)(2 * lane + 1) * (1.f / 64.f));
  const float c0 = __cosf(a0), s0 = __sinf(a0);
  const float c1 = __cosf(a1), s1 = __sinf(a1);
  // out[2i] = x_odd*cos - x_even*sin ; out[2i+1] = x_odd*sin + x_even*cos
  const float qe0 = q.y * c0 - q.x * s0, qo0 = q.y * s0 + q.x * c0;
  const float qe1 = q.w * c1 - q.z * s1, qo1 = q.w * s1 + q.z * c1;
  const float ke0 = k.y * c0 - k.x * s0, ko0 = k.y * s0 + k.x * c0;
  const float ke1 = k.w * c1 - k.z * s1, ko1 = k.w * s1 + k.z * c1;

  const size_t qoff = ((size_t)h * SEQ + s) * HD + lane * 4;
  qb[qoff + 0] = (bf16)qe0; qb[qoff + 1] = (bf16)qo0;
  qb[qoff + 2] = (bf16)qe1; qb[qoff + 3] = (bf16)qo1;
  kb[qoff + 0] = (bf16)ke0; kb[qoff + 1] = (bf16)ko0;
  kb[qoff + 2] = (bf16)ke1; kb[qoff + 3] = (bf16)ko1;

  const size_t vtoff = ((size_t)h * HD + lane * 4) * SEQ + s;
  vt[vtoff + 0 * SEQ] = (bf16)v.x;
  vt[vtoff + 1 * SEQ] = (bf16)v.y;
  vt[vtoff + 2 * SEQ] = (bf16)v.z;
  vt[vtoff + 3 * SEQ] = (bf16)v.w;
}

// ---------------- Flash attention (mask: q<=kv & (kv%16==0 | kv>=q-16)) ----
// grid = (SEQ/128 q-blocks, NH heads); wave owns 16 query rows; TDM double-
// buffers the 32-wide K/V tiles.

__global__ __launch_bounds__(256) void attn_kernel(const bf16* __restrict__ qb,
                                                   const bf16* __restrict__ kb,
                                                   const bf16* __restrict__ vt,
                                                   bf16* __restrict__ o) {
  __shared__ bf16 Kl[2][32][128];   // [kv][d]
  __shared__ bf16 Vl[2][128][32];   // [d][kv] (transposed V tile)
  __shared__ bf16 Pl[8][16][32];    // per-wave P scratch [qrow][kv]

  const int qblk = blockIdx.x, h = blockIdx.y;
  const int tid = threadIdx.x, lane = tid & 31, wave = tid >> 5;
  const int g = lane >> 4, ln = lane & 15;
  const int q0 = qblk * 128 + wave * 16;

  v16bf qf[4];
  const bf16* qrow = qb + ((size_t)h * SEQ + q0 + ln) * HD;
#pragma unroll
  for (int kt = 0; kt < 4; ++kt) qf[kt] = frag_a(qrow + kt * 32, g);

  v8f oacc[8];
#pragma unroll
  for (int i = 0; i < 8; ++i)
#pragma unroll
    for (int e = 0; e < 8; ++e) oacc[i][e] = 0.f;
  float mrow[8], lrow[8];
#pragma unroll
  for (int r = 0; r < 8; ++r) { mrow[r] = -1e30f; lrow[r] = 0.f; }

  if (tid == 0) {
    tdm_load_2d(lds_off(&Kl[0][0][0]),
                kb + ((size_t)h * SEQ + qblk * 128) * HD, 32, 128, HD);
    tdm_load_2d(lds_off(&Vl[0][0][0]),
                vt + (size_t)h * HD * SEQ + qblk * 128, 128, 32, SEQ);
  }

  int buf = 0;
  for (int kv0 = qblk * 128; kv0 < SEQ; kv0 += 32) {
    if (tid == 0) {
      if (kv0 + 32 < SEQ) {
        tdm_load_2d(lds_off(&Kl[buf ^ 1][0][0]),
                    kb + ((size_t)h * SEQ + kv0 + 32) * HD, 32, 128, HD);
        tdm_load_2d(lds_off(&Vl[buf ^ 1][0][0]),
                    vt + (size_t)h * HD * SEQ + (kv0 + 32), 128, 32, SEQ);
        __builtin_amdgcn_s_wait_tensorcnt(2);
      } else {
        __builtin_amdgcn_s_wait_tensorcnt(0);
      }
    }
    __syncthreads();

    // S = Q @ K^T
    v8f sacc[2];
#pragma unroll
    for (int t = 0; t < 2; ++t) {
#pragma unroll
      for (int e = 0; e < 8; ++e) sacc[t][e] = 0.f;
#pragma unroll
      for (int kt = 0; kt < 4; ++kt) {
        const v16bf bfr = frag_b(&Kl[buf][t * 16 + ln][kt * 32], g);
        sacc[t] = wmma_bf16(qf[kt], bfr, sacc[t]);
      }
    }

    // Online softmax per query row.
#pragma unroll
    for (int r = 0; r < 8; ++r) {
      const int qg = q0 + r + 8 * g;
      float s0v = sacc[0][r] * ATT_SCALE;
      float s1v = sacc[1][r] * ATT_SCALE;
      const int kva = kv0 + ln, kvb = kv0 + 16 + ln;
      const bool keepa = (qg <= kva) && (((kva & 15) == 0) || (kva >= qg - 16));
      const bool keepb = (qg <= kvb) && (((kvb & 15) == 0) || (kvb >= qg - 16));
      if (!keepa) s0v = -1e30f;
      if (!keepb) s1v = -1e30f;

      float mx = fmaxf(s0v, s1v);
#pragma unroll
      for (int off = 1; off < 16; off <<= 1) mx = fmaxf(mx, __shfl_xor(mx, off, 32));
      const float mnew = fmaxf(mrow[r], mx);
      const float scl = __expf(mrow[r] - mnew);
      const float p0 = __expf(s0v - mnew);
      const float p1 = __expf(s1v - mnew);
      float sum = p0 + p1;
#pragma unroll
      for (int off = 1; off < 16; off <<= 1) sum += __shfl_xor(sum, off, 32);
      lrow[r] = lrow[r] * scl + sum;
      mrow[r] = mnew;
#pragma unroll
      for (int i = 0; i < 8; ++i) oacc[i][r] *= scl;
      Pl[wave][r + 8 * g][ln] = (bf16)p0;
      Pl[wave][r + 8 * g][16 + ln] = (bf16)p1;
    }
    __syncthreads();

    // O += P @ V
    const v16bf pf = frag_a(&Pl[wave][ln][0], g);
#pragma unroll
    for (int i = 0; i < 8; ++i) {
      const v16bf bfr = frag_b(&Vl[buf][i * 16 + ln][0], g);
      oacc[i] = wmma_bf16(pf, bfr, oacc[i]);
    }
    __syncthreads();
    buf ^= 1;
  }

#pragma unroll
  for (int i = 0; i < 8; ++i)
#pragma unroll
    for (int r = 0; r < 8; ++r) {
      const int row = q0 + r + 8 * g;
      const int col = h * HD + i * 16 + ln;
      o[(size_t)row * HID + col] = (bf16)(oacc[i][r] / lrow[r]);
    }
}

// ---------------------------------------------------------------------------
// Host-side orchestration.
// ---------------------------------------------------------------------------

extern "C" void kernel_launch(void* const* d_in, const int* in_sizes, int n_in,
                              void* d_out, int out_size, void* d_ws, size_t ws_size,
                              hipStream_t stream) {
  (void)in_sizes; (void)n_in; (void)out_size; (void)ws_size;

  const int*   ids    = (const int*)d_in[0];
  const float* wte    = (const float*)d_in[1];
  const float* qkv_wf = (const float*)d_in[2];
  const float* out_wf = (const float*)d_in[3];
  const float* w1f    = (const float*)d_in[4];
  const float* w2f    = (const float*)d_in[5];
  const float* lm_wf  = (const float*)d_in[6];
  float* logits = (float*)d_out;

  const size_t W_QKV = (size_t)NLAYER * 3 * HID * HID;
  const size_t W_OUT = (size_t)NLAYER * HID * HID;
  const size_t W_W1  = (size_t)NLAYER * 4 * HID * HID;
  const size_t W_W2  = (size_t)NLAYER * 4 * HID * HID;
  const size_t W_LM  = (size_t)VOCAB * HID;

  char* ws = (char*)d_ws;
  size_t off = 0;
  auto take = [&](size_t bytes) { size_t o = off; off += (bytes + 255) & ~(size_t)255; return o; };

  float* x    = (float*)(ws + take((size_t)SEQ * HID * 4));
  bf16*  hbuf = (bf16*)(ws + take((size_t)SEQ * HID * 2));
  float* qkv  = (float*)(ws + take((size_t)SEQ * 3 * HID * 4));
  bf16*  obuf = (bf16*)(ws + take((size_t)SEQ * HID * 2));
  bf16*  gact = (bf16*)(ws + take((size_t)SEQ * 4 * HID * 2));
  bf16*  xb   = (bf16*)(ws + take((size_t)SEQ * HID * 2));
  bf16*  qb   = (bf16*)(ws + take((size_t)SEQ * HID * 2));
  bf16*  kb   = (bf16*)(ws + take((size_t)SEQ * HID * 2));
  bf16*  vtb  = (bf16*)(ws + take((size_t)SEQ * HID * 2));
  bf16*  qkv_w = (bf16*)(ws + take(W_QKV * 2));
  bf16*  out_w = (bf16*)(ws + take(W_OUT * 2));
  bf16*  w1   = (bf16*)(ws + take(W_W1 * 2));
  bf16*  w2   = (bf16*)(ws + take(W_W2 * 2));
  bf16*  lm_w = (bf16*)(ws + take(W_LM * 2));

  auto cvt = [&](const float* s, bf16* d, size_t n) {
    const int n4 = (int)(n / 4);
    cvt_kernel<<<(n4 + 255) / 256, 256, 0, stream>>>(s, d, n4);
  };

  // One-time weight conversion to bf16 (halves weight-side memory traffic).
  cvt(qkv_wf, qkv_w, W_QKV);
  cvt(out_wf, out_w, W_OUT);
  cvt(w1f, w1, W_W1);
  cvt(w2f, w2, W_W2);
  cvt(lm_wf, lm_w, W_LM);

  // x = rms(wte[input_ids])
  rms_kernel<true, float><<<SEQ, 256, 0, stream>>>(ids, wte, x);

  for (int l = 0; l < NLAYER; ++l) {
    // h = rms(x) in bf16 ; qkv = h @ Wqkv^T
    rms_kernel<false, bf16><<<SEQ, 256, 0, stream>>>(nullptr, x, hbuf);
    gemm_kernel<false, false, float><<<dim3(3 * HID / 128, SEQ / 128), 256, 0, stream>>>(
        hbuf, qkv_w + (size_t)l * 3 * HID * HID, nullptr, qkv, SEQ, 3 * HID, HID);

    // q/k rms + rope -> bf16, V transposed -> bf16
    qkprep_kernel<<<SEQ, 256, 0, stream>>>(qkv, qb, kb, vtb);

    // o = attention(q, k, v)  (bf16 output)
    attn_kernel<<<dim3(SEQ / 128, NH), 256, 0, stream>>>(qb, kb, vtb, obuf);

    // x = x + o @ Wo^T
    gemm_kernel<true, false, float><<<dim3(HID / 128, SEQ / 128), 256, 0, stream>>>(
        obuf, out_w + (size_t)l * HID * HID, x, x, SEQ, HID, HID);

    // g = gelu(rms(x) @ W1^T) in bf16 ; x = x + g @ W2^T
    rms_kernel<false, bf16><<<SEQ, 256, 0, stream>>>(nullptr, x, hbuf);
    gemm_kernel<false, true, bf16><<<dim3(4 * HID / 128, SEQ / 128), 256, 0, stream>>>(
        hbuf, w1 + (size_t)l * 4 * HID * HID, nullptr, gact, SEQ, 4 * HID, HID);
    gemm_kernel<true, false, float><<<dim3(HID / 128, SEQ / 128), 256, 0, stream>>>(
        gact, w2 + (size_t)l * HID * 4 * HID, x, x, SEQ, HID, 4 * HID);
  }

  // logits = x @ lm_head^T (raw residual stream, no final rms in reference)
  cvt(x, xb, (size_t)SEQ * HID);
  gemm_kernel<false, false, float><<<dim3(VOCAB / 128, SEQ / 128), 256, 0, stream>>>(
      xb, lm_w, nullptr, logits, SEQ, VOCAB, HID);
}
